// IntegralTransform_66803921322478
// MI455X (gfx1250) — matrix operand order repaired and checked by reading, not compiled, not run on hardware
//
#include <hip/hip_runtime.h>
#include <hip/hip_bf16.h>
#include <string.h>

typedef _Float16 half_t;
typedef __attribute__((ext_vector_type(16))) _Float16 v16h;
typedef __attribute__((ext_vector_type(8)))  float    v8f;

// LDS row pitches (halves). Odd-dword strides -> conflict-free ds_load_b32 column walks.
#define LDA_AGG 98    // 96 used (70 real + pad to K multiple of 32)
#define LDA_H   130   // 128 used

// Weight-fragment workspace layout (halves): frag = 32 lanes x 16 halves = 512
#define W1_FRAGS 24   // 3 ktiles x 8 ntiles
#define W2_FRAGS 32   // 4 ktiles x 8 ntiles
#define W3_FRAGS 16   // 4 ktiles x 4 ntiles
#define W1_OFF   0
#define W2_OFF   (W1_FRAGS * 512)
#define W3_OFF   ((W1_FRAGS + W2_FRAGS) * 512)
#define TOTAL_WS_HALVES ((W1_FRAGS + W2_FRAGS + W3_FRAGS) * 512)

// ---------------------------------------------------------------------------
// Pack f32 weights into f16 WMMA B-fragment layout:
//   B tile (32K x 16N): lane l -> n = l&15, k_local = (h<8 ? (l>>4)*8 + h : 16 + (l>>4)*8 + h-8), h = 0..15
// ---------------------------------------------------------------------------
__global__ __launch_bounds__(256) void it_prep_weights(
    const float* __restrict__ W1, const float* __restrict__ W2,
    const float* __restrict__ W3, half_t* __restrict__ ws)
{
    int tid = blockIdx.x * blockDim.x + threadIdx.x;
    if (tid >= TOTAL_WS_HALVES) return;
    int f    = tid >> 9;         // fragment index
    int r    = tid & 511;
    int lane = r >> 4;
    int h    = r & 15;
    int kl   = (h < 8) ? (((lane >> 4) << 3) + h) : (16 + ((lane >> 4) << 3) + (h - 8));   // 0..31 within K-tile
    int nl   = lane & 15;
    float v = 0.0f;
    if (f < W1_FRAGS) {                  // W1: [70 x 128], K padded to 96
        int kt = f >> 3, nt = f & 7;
        int k = kt * 32 + kl, n = nt * 16 + nl;
        if (k < 70) v = W1[k * 128 + n];
    } else if (f < W1_FRAGS + W2_FRAGS) { // W2: [128 x 128]
        int f2 = f - W1_FRAGS;
        int kt = f2 >> 3, nt = f2 & 7;
        v = W2[(kt * 32 + kl) * 128 + (nt * 16 + nl)];
    } else {                              // W3: [128 x 64]
        int f3 = f - (W1_FRAGS + W2_FRAGS);
        int kt = f3 >> 2, nt = f3 & 3;
        v = W3[(kt * 32 + kl) * 64 + (nt * 16 + nl)];
    }
    ws[tid] = (half_t)v;
}

// A fragment (16M x 32K f16) from an LDS row-major half array.
// lanes 0-15: M=lane, V0..V3 hold K=0..7 pairs, V4..V7 hold K=16..23
// lanes16-31: M=lane-16, V0..V3 K=8..15, V4..V7 K=24..31
__device__ __forceinline__ v16h load_a_frag(const half_t* base, int ld,
                                            int mt, int kt, int lane)
{
    int m  = mt * 16 + (lane & 15);
    int kh = (lane >> 4) * 8;
    const unsigned* rowu = (const unsigned*)(base + m * ld + kt * 32);
    union { v16h v; unsigned u[8]; } t;
#pragma unroll
    for (int r = 0; r < 8; ++r) {
        int k0 = ((r < 4) ? (2 * r) : (16 + 2 * (r - 4))) + kh; // even
        t.u[r] = rowu[k0 >> 1];
    }
    return t.v;
}

__device__ __forceinline__ v8f wmma_f16(v16h a, v16h b, v8f c)
{
    return __builtin_amdgcn_wmma_f32_16x16x32_f16(
        /*neg_a=*/false, a, /*neg_b=*/false, b,
        /*c_mod=*/(short)0, c, /*reuse_a=*/false, /*reuse_b=*/false);
}

__device__ __forceinline__ float gelu_erf(float x)
{
    return 0.5f * x * (1.0f + erff(x * 0.70710678118654752440f));
}

// ---------------------------------------------------------------------------
// One workgroup (256 thr = 8 waves) handles one output point (chunks of 32 edges)
// ---------------------------------------------------------------------------
__global__ __launch_bounds__(256) void it_mlp_kernel(
    const float* __restrict__ y, const float* __restrict__ f_y,
    const int* __restrict__ nbr, const int* __restrict__ rs,
    const float* __restrict__ b1, const float* __restrict__ b2,
    const float* __restrict__ b3, const half_t* __restrict__ wfrag,
    float* __restrict__ out, int n_points)
{
    __shared__ half_t agg[32 * LDA_AGG];
    __shared__ half_t h1 [32 * LDA_H];
    __shared__ half_t h2 [32 * LDA_H];
    __shared__ int    nidx[32];
    __shared__ float  osum[64];

    int i = blockIdx.x;
    if (i >= n_points) return;
    int tid  = threadIdx.x;
    int lane = tid & 31;
    int wave = tid >> 5;

    int start = rs[i], end = rs[i + 1];
    int count = end - start;

    if (tid < 64) osum[tid] = 0.0f;

    float yi0 = y[3 * i + 0], yi1 = y[3 * i + 1], yi2 = y[3 * i + 2];

    for (int e0 = start; e0 < end; e0 += 32) {
        int valid = end - e0; if (valid > 32) valid = 32;

        if (tid < 32) nidx[tid] = (tid < valid) ? nbr[e0 + tid] : 0;
        __syncthreads();

        // ---- stage agg[32 x 96] = [rep_y(3) | self_y(3) | f_y(64) | pad] ----
        for (int idx = tid; idx < 32 * 96; idx += 256) {
            int m = idx / 96, k = idx - m * 96;
            float v = 0.0f;
            if (m < valid) {
                int nj = nidx[m];
                if      (k < 3)  v = y[3 * nj + k];
                else if (k < 6)  v = (k == 3) ? yi0 : (k == 4) ? yi1 : yi2;
                else if (k < 70) v = f_y[64 * nj + (k - 6)];
            }
            agg[m * LDA_AGG + k] = (half_t)v;
        }
        __syncthreads();

        // ---- layer 1: [32x96] @ W1[96x128] + b1, gelu -> h1 ----
#pragma unroll
        for (int jj = 0; jj < 2; ++jj) {
            int j = wave * 2 + jj;
            int mt = j & 1, nt = j >> 1;
            v8f c = {};
#pragma unroll
            for (int kt = 0; kt < 3; ++kt) {
                v16h a = load_a_frag(agg, LDA_AGG, mt, kt, lane);
                v16h b = *(const v16h*)(wfrag + W1_OFF + ((kt * 8 + nt) * 32 + lane) * 16);
                c = wmma_f16(a, b, c);
            }
            int n = nt * 16 + (lane & 15);
            float bias = b1[n];
            int mbase = mt * 16 + ((lane >> 4) ? 8 : 0);
#pragma unroll
            for (int r = 0; r < 8; ++r) {
                float x = c[r] + bias;
                h1[(mbase + r) * LDA_H + n] = (half_t)gelu_erf(x);
            }
        }
        __syncthreads();

        // ---- layer 2: h1[32x128] @ W2[128x128] + b2, gelu -> h2 ----
#pragma unroll
        for (int jj = 0; jj < 2; ++jj) {
            int j = wave * 2 + jj;
            int mt = j & 1, nt = j >> 1;
            v8f c = {};
#pragma unroll
            for (int kt = 0; kt < 4; ++kt) {
                v16h a = load_a_frag(h1, LDA_H, mt, kt, lane);
                v16h b = *(const v16h*)(wfrag + W2_OFF + ((kt * 8 + nt) * 32 + lane) * 16);
                c = wmma_f16(a, b, c);
            }
            int n = nt * 16 + (lane & 15);
            float bias = b2[n];
            int mbase = mt * 16 + ((lane >> 4) ? 8 : 0);
#pragma unroll
            for (int r = 0; r < 8; ++r) {
                float x = c[r] + bias;
                h2[(mbase + r) * LDA_H + n] = (half_t)gelu_erf(x);
            }
        }
        __syncthreads();

        // ---- layer 3: h2[32x128] @ W3[128x64] + b3, masked column sums ----
        {
            int mt = wave & 1, nt = wave >> 1;     // 8 waves -> 2 x 4 tile jobs
            v8f c = {};
#pragma unroll
            for (int kt = 0; kt < 4; ++kt) {
                v16h a = load_a_frag(h2, LDA_H, mt, kt, lane);
                v16h b = *(const v16h*)(wfrag + W3_OFF + ((kt * 4 + nt) * 32 + lane) * 16);
                c = wmma_f16(a, b, c);
            }
            int n = nt * 16 + (lane & 15);
            float bias = b3[n];
            int mbase = mt * 16 + ((lane >> 4) ? 8 : 0);
            float s = 0.0f;
#pragma unroll
            for (int r = 0; r < 8; ++r)
                if (mbase + r < valid) s += c[r] + bias;
            atomicAdd(&osum[n], s);   // ds_add_f32
        }
        __syncthreads();
    }

    if (tid < 64) {
        int cnt = (count > 1) ? count : 1;
        out[i * 64 + tid] = osum[tid] * (1.0f / (float)cnt);
    }
}

// ---------------------------------------------------------------------------
extern "C" void kernel_launch(void* const* d_in, const int* in_sizes, int n_in,
                              void* d_out, int out_size, void* d_ws, size_t ws_size,
                              hipStream_t stream)
{
    const float* y   = (const float*)d_in[0];
    const float* f_y = (const float*)d_in[1];
    const int*   nbr = (const int*)  d_in[2];
    const int*   rs  = (const int*)  d_in[3];
    const float* W1  = (const float*)d_in[4];
    const float* b1  = (const float*)d_in[5];
    const float* W2  = (const float*)d_in[6];
    const float* b2  = (const float*)d_in[7];
    const float* W3  = (const float*)d_in[8];
    const float* b3  = (const float*)d_in[9];

    int n_points = in_sizes[3] - 1;   // row_splits has N+1 entries
    half_t* ws = (half_t*)d_ws;

    int prep_blocks = (TOTAL_WS_HALVES + 255) / 256;
    it_prep_weights<<<dim3(prep_blocks), dim3(256), 0, stream>>>(W1, W2, W3, ws);

    it_mlp_kernel<<<dim3(n_points), dim3(256), 0, stream>>>(
        y, f_y, nbr, rs, b1, b2, b3, ws, (float*)d_out, n_points);
}
